// Encoder_69329362092650
// MI455X (gfx1250) — matrix operand order repaired and checked by reading, compiled, not run on hardware
//
#include <hip/hip_runtime.h>
#include <math.h>

// ---------------------------------------------------------------------------
// Problem: BATCH=128, T=128, N=128, M=256.  Attention-gated LSTM scan.
//   Precompute Ux[b,n,s] = sum_t x[b,t,n]*Ue[s,t]         (WMMA bf16)
//   Scan over t: wp = [h,c]@We^T ; z=tanh(wp+Ux); scores=z.ve; softmax;
//                gates = [x_t*alpha, h]@[W_ih|W_hh]^T + bias; LSTM cell.
// Design: 8 persistent WGs x 256 threads (8 waves), 16 batches per WG,
// everything L2-resident, c-state in accumulator-layout registers.
// CDNA5 paths: v_wmma_f32_16x16x32_bf16, global_load_async_to_lds_b128 +
// s_wait_asynccnt, global_prefetch_b8, ds_load_b128 A-fragments.
// ---------------------------------------------------------------------------

typedef __bf16        bf16x16 __attribute__((ext_vector_type(16)));
typedef float         f32x8   __attribute__((ext_vector_type(8)));
typedef unsigned int  u32x4   __attribute__((ext_vector_type(4)));
typedef unsigned short u16;

union Frag16 { bf16x16 v; u32x4 q[2]; };

static __device__ __forceinline__ u16 f2bf(float f) {
  unsigned int u = __float_as_uint(f);
  u += 0x7FFFu + ((u >> 16) & 1u);           // round-to-nearest-even
  return (u16)(u >> 16);
}
static __device__ __forceinline__ float sigm(float v) {
  return 1.0f / (1.0f + expf(-v));
}

// ---------------------------------------------------------------------------
// Prep: bf16 weight packing, bias fuse, x transpose (B,T,N)->(B,N,T) bf16.
// ---------------------------------------------------------------------------
__global__ __launch_bounds__(256) void prep_kernel(
    const float* __restrict__ x, const float* __restrict__ We,
    const float* __restrict__ Ue, const float* __restrict__ W_ih,
    const float* __restrict__ W_hh, const float* __restrict__ b_ih,
    const float* __restrict__ b_hh,
    u16* __restrict__ We16, u16* __restrict__ Ue16, u16* __restrict__ Wg16,
    u16* __restrict__ xT16, float* __restrict__ bias) {
  const int stride = gridDim.x * blockDim.x;
  const int tid0 = blockIdx.x * blockDim.x + threadIdx.x;
  for (int i = tid0; i < 128 * 512; i += stride) We16[i] = f2bf(We[i]);
  for (int i = tid0; i < 128 * 128; i += stride) Ue16[i] = f2bf(Ue[i]);
  for (int i = tid0; i < 1024; i += stride) bias[i] = b_ih[i] + b_hh[i];
  for (int i = tid0; i < 1024 * 384; i += stride) {
    int n = i / 384, k = i - n * 384;
    float v = (k < 128) ? W_ih[n * 128 + k] : W_hh[n * 256 + (k - 128)];
    Wg16[i] = f2bf(v);
  }
  for (long long i = tid0; i < (long long)128 * 128 * 128; i += stride) {
    int b = (int)(i >> 14);
    int r = (int)(i & 16383);
    int n = r >> 7, tt = r & 127;
    xT16[i] = f2bf(x[((long long)b * 128 + tt) * 128 + n]);
  }
}

// ---------------------------------------------------------------------------
// Ux[b,n,s] = (x[b]^T @ Ue^T)[n,s].  One 16x16 tile per wave, K=128 (4 WMMA).
// A = xT16[b] row-major (n,t); B^T = Ue row-major (s,t) -> contiguous loads.
// ---------------------------------------------------------------------------
__global__ __launch_bounds__(256) void ux_kernel(
    const u16* __restrict__ xT16, const u16* __restrict__ Ue16,
    float* __restrict__ Ux) {
  const int wid = (blockIdx.x * 256 + threadIdx.x) >> 5;
  const int lane = threadIdx.x & 31;
  const int half = lane >> 4, nL = lane & 15;
  const int b = wid >> 6;
  const int rest = wid & 63;
  const int ntile = rest >> 3, stile = rest & 7;

  const u16* arow = xT16 + ((size_t)b * 128 + ntile * 16 + nL) * 128;
  const u16* brow = Ue16 + (size_t)(stile * 16 + nL) * 128;
  f32x8 acc = {};
#pragma unroll
  for (int kt = 0; kt < 4; ++kt) {
    Frag16 a, bf;
    const u16* ap = arow + kt * 32 + half * 8;
    a.q[0] = *(const u32x4*)ap;
    a.q[1] = *(const u32x4*)(ap + 16);
    const u16* bp = brow + kt * 32 + half * 8;
    bf.q[0] = *(const u32x4*)bp;
    bf.q[1] = *(const u32x4*)(bp + 16);
    acc = __builtin_amdgcn_wmma_f32_16x16x32_bf16(false, a.v, false, bf.v,
                                                  (short)0, acc, false, false);
  }
  float* orow = Ux + ((size_t)b * 128 + ntile * 16) * 128 + stile * 16 + nL;
#pragma unroll
  for (int r = 0; r < 8; ++r) orow[(size_t)(r + 8 * half) * 128] = acc[r];
}

// ---------------------------------------------------------------------------
// Persistent scan kernel: blockIdx.x owns batches [16*bid, 16*bid+16).
// ---------------------------------------------------------------------------
__global__ __launch_bounds__(256) void scan_kernel(
    const float* __restrict__ x, const float* __restrict__ v_e,
    const float* __restrict__ bias, const u16* __restrict__ We16,
    const u16* __restrict__ Wg16, const float* __restrict__ Ux,
    float* __restrict__ out) {
  __shared__ u16 hs16[16][520];   // [bb][0:256)=h, [256:512)=c (bf16), +8 pad
  __shared__ u16 xg16[16][392];   // [bb][0:128)=xw, [128:384)=h (bf16), +8 pad
  __shared__ float wp[16][132];   // hs @ We^T  (16 x T)
  __shared__ float sc[16][129];   // scores -> exp(scores)
  __shared__ float xS[16][128];   // async-staged x_t tile (8 KB)
  __shared__ float part[16][16];
  __shared__ float redM[16], redS[16];
  __shared__ float veS[128];

  const int tid = threadIdx.x;
  const int lane = tid & 31, wave = tid >> 5;
  const int half = lane >> 4, nL = lane & 15;
  const int batch0 = blockIdx.x * 16;

  for (int i = tid; i < 16 * 520; i += 256) ((u16*)hs16)[i] = 0;
  for (int i = tid; i < 16 * 392; i += 256) ((u16*)xg16)[i] = 0;
  if (tid < 128) veS[tid] = v_e[tid];

  // Per-thread async-staging addresses (constant across steps except t).
  const int sbb = tid >> 4;            // batch row this thread stages
  const int sn8 = (tid & 15) * 8;      // 8 floats = two b128 chunks
  const unsigned ldsX =
      (unsigned)(size_t)&xS[sbb][sn8];  // AS3 offset (low 32 bits of generic)
  const float* gX0 =
      x + ((size_t)(batch0 + sbb) * 128) * 128 + sn8;  // + t*128 per step

  f32x8 cfr[2];  // c-state, accumulator layout: 2 sub-tiles x 8 rows
  cfr[0] = (f32x8){};
  cfr[1] = (f32x8){};
  __syncthreads();

  for (int t = 0; t < 128; ++t) {
    // ---- Async stage x_t -> LDS; drains behind Phase 1+2 compute. ----
    {
      unsigned long long ga = (unsigned long long)(size_t)(gX0 + (size_t)t * 128);
      asm volatile("global_load_async_to_lds_b128 %0, %1, off"
                   :: "v"(ldsX), "v"(ga) : "memory");
      asm volatile("global_load_async_to_lds_b128 %0, %1, off offset:16"
                   :: "v"(ldsX), "v"(ga) : "memory");
    }

    // ---- Phase 1: wp = [h,c] @ We^T.  Wave w -> time cols [16w,16w+16). ----
    {
      f32x8 acc = {};
      const u16* wRow = We16 + (size_t)(wave * 16 + nL) * 512;
#pragma unroll
      for (int kt = 0; kt < 16; ++kt) {
        if (kt + 1 < 16) __builtin_prefetch(wRow + (kt + 1) * 32, 0, 0);
        Frag16 a, b;
        const u16* ap = &hs16[nL][kt * 32 + half * 8];
        a.q[0] = *(const u32x4*)ap;
        a.q[1] = *(const u32x4*)(ap + 16);
        const u16* bp = wRow + kt * 32 + half * 8;
        b.q[0] = *(const u32x4*)bp;
        b.q[1] = *(const u32x4*)(bp + 16);
        acc = __builtin_amdgcn_wmma_f32_16x16x32_bf16(
            false, a.v, false, b.v, (short)0, acc, false, false);
      }
#pragma unroll
      for (int r = 0; r < 8; ++r) wp[r + 8 * half][wave * 16 + nL] = acc[r];
    }
    __syncthreads();

    // ---- Phase 2: scores[bb,n] = sum_t ve[t]*tanh(wp[bb,t]+Ux[b,n,t]). ----
    for (int j = 0; j < 8; ++j) {
      const int p = tid + 256 * j;
      const int bb = p >> 7, n = p & 127;
      const float* uxp = Ux + ((size_t)(batch0 + bb) * 128 + n) * 128;
      float s = 0.f;
#pragma unroll 4
      for (int t4 = 0; t4 < 32; ++t4) {
        float4 u = *(const float4*)(uxp + 4 * t4);
        const float* wpr = &wp[bb][4 * t4];
        const float* vep = &veS[4 * t4];
        s += vep[0] * tanhf(wpr[0] + u.x);
        s += vep[1] * tanhf(wpr[1] + u.y);
        s += vep[2] * tanhf(wpr[2] + u.z);
        s += vep[3] * tanhf(wpr[3] + u.w);
      }
      sc[bb][n] = s;
    }
    __syncthreads();

    // ---- Softmax over n (two-level reduction). ----
    {
      const int bb = tid >> 4, li = tid & 15;
      float m = -1e30f;
      for (int q = 0; q < 8; ++q) m = fmaxf(m, sc[bb][li * 8 + q]);
      part[bb][li] = m;
      __syncthreads();
      if (li == 0) {
        float mm = part[bb][0];
        for (int q = 1; q < 16; ++q) mm = fmaxf(mm, part[bb][q]);
        redM[bb] = mm;
      }
      __syncthreads();
      const float mv = redM[bb];
      float e = 0.f;
      for (int q = 0; q < 8; ++q) {
        float ev = expf(sc[bb][li * 8 + q] - mv);
        sc[bb][li * 8 + q] = ev;
        e += ev;
      }
      part[bb][li] = e;
      __syncthreads();
      if (li == 0) {
        float ss = 0.f;
        for (int q = 0; q < 16; ++q) ss += part[bb][q];
        redS[bb] = 1.0f / ss;
      }
    }
    // Drain this wave's async x_t stores; barrier publishes them to the WG.
    asm volatile("s_wait_asynccnt 0x0" ::: "memory");
    __syncthreads();

    // ---- Phase 2.5: xw = x_t * alpha -> bf16 into xg16[:,0:128). ----
    for (int j = 0; j < 8; ++j) {
      const int p = tid + 256 * j;
      const int bb = p >> 7, n = p & 127;
      xg16[bb][n] = f2bf(xS[bb][n] * sc[bb][n] * redS[bb]);
    }
    __syncthreads();

    // ---- Phase 3: gates = [xw,h] @ [W_ih|W_hh]^T.  Wave w owns feature
    // cols [32w,32w+32) of each gate -> i,f,g,o all in-register per lane. ----
    f32x8 gacc[4][2];
#pragma unroll
    for (int gi = 0; gi < 4; ++gi) {
#pragma unroll
      for (int sub = 0; sub < 2; ++sub) {
        f32x8 acc = {};
        const u16* wr =
            Wg16 + (size_t)(gi * 256 + wave * 32 + sub * 16 + nL) * 384;
#pragma unroll
        for (int kt = 0; kt < 12; ++kt) {
          if (kt + 1 < 12) __builtin_prefetch(wr + (kt + 1) * 32, 0, 0);
          Frag16 a, b;
          const u16* ap = &xg16[nL][kt * 32 + half * 8];
          a.q[0] = *(const u32x4*)ap;
          a.q[1] = *(const u32x4*)(ap + 16);
          const u16* bp = wr + kt * 32 + half * 8;
          b.q[0] = *(const u32x4*)bp;
          b.q[1] = *(const u32x4*)(bp + 16);
          acc = __builtin_amdgcn_wmma_f32_16x16x32_bf16(
              false, a.v, false, b.v, (short)0, acc, false, false);
        }
        gacc[gi][sub] = acc;
      }
    }
    __syncthreads();  // all waves done reading xg16 before h-section rewrite

    // ---- LSTM cell (registers only for c), write h to out/LDS. ----
#pragma unroll
    for (int sub = 0; sub < 2; ++sub) {
      const int mfeat = wave * 32 + sub * 16 + nL;
      const float bi = bias[0 * 256 + mfeat];
      const float bfv = bias[1 * 256 + mfeat];
      const float bg = bias[2 * 256 + mfeat];
      const float bo = bias[3 * 256 + mfeat];
#pragma unroll
      for (int r = 0; r < 8; ++r) {
        const int bb = 8 * half + r;
        const float gi_ = gacc[0][sub][r] + bi;
        const float gf_ = gacc[1][sub][r] + bfv;
        const float gg_ = gacc[2][sub][r] + bg;
        const float go_ = gacc[3][sub][r] + bo;
        const float cc = cfr[sub][r];
        const float c2 = sigm(gf_) * cc + sigm(gi_) * tanhf(gg_);
        const float h2 = sigm(go_) * tanhf(c2);
        cfr[sub][r] = c2;
        out[((size_t)t * 128 + (batch0 + bb)) * 256 + mfeat] = h2;
        const u16 hb = f2bf(h2), cb = f2bf(c2);
        hs16[bb][mfeat] = hb;
        hs16[bb][256 + mfeat] = cb;
        xg16[bb][128 + mfeat] = hb;
      }
    }
    __syncthreads();  // publish new h/c before next step's Phase 1
  }
}

// ---------------------------------------------------------------------------
extern "C" void kernel_launch(void* const* d_in, const int* in_sizes, int n_in,
                              void* d_out, int out_size, void* d_ws,
                              size_t ws_size, hipStream_t stream) {
  const float* x = (const float*)d_in[0];
  const float* We = (const float*)d_in[1];
  const float* Ue = (const float*)d_in[2];
  const float* v_e = (const float*)d_in[3];
  const float* W_ih = (const float*)d_in[4];
  const float* W_hh = (const float*)d_in[5];
  const float* b_ih = (const float*)d_in[6];
  const float* b_hh = (const float*)d_in[7];
  float* out = (float*)d_out;

  char* ws = (char*)d_ws;
  float* Ux  = (float*)(ws);                                   // 8 MB
  u16* xT16  = (u16*)(ws + (size_t)(8u << 20));                // 4 MB
  u16* Wg16  = (u16*)(ws + (size_t)(12u << 20));               // 768 KB
  u16* We16  = (u16*)(ws + (size_t)(13u << 20));               // 128 KB
  u16* Ue16  = (u16*)(ws + (size_t)(13u << 20) + (256u << 10));// 32 KB
  float* bias = (float*)(ws + (size_t)(13u << 20) + (512u << 10));

  prep_kernel<<<512, 256, 0, stream>>>(x, We, Ue, W_ih, W_hh, b_ih, b_hh,
                                       We16, Ue16, Wg16, xT16, bias);
  ux_kernel<<<1024, 256, 0, stream>>>(xT16, Ue16, Ux);
  scan_kernel<<<8, 256, 0, stream>>>(x, v_e, bias, We16, Wg16, Ux, out);
}
